// simple_model_11819749998948
// MI455X (gfx1250) — matrix operand order repaired and checked by reading, not compiled, not run on hardware
//
#include <hip/hip_runtime.h>

typedef __attribute__((ext_vector_type(16))) __bf16 v16bf;
typedef __attribute__((ext_vector_type(8)))  float  v8f;
typedef __attribute__((ext_vector_type(4)))  unsigned v4u;
typedef __attribute__((ext_vector_type(8)))  int      v8i;
typedef __attribute__((ext_vector_type(4)))  int      v4i;

#if defined(__has_builtin)
#if __has_builtin(__builtin_amdgcn_tensor_load_to_lds) && __has_builtin(__builtin_amdgcn_s_wait_tensorcnt)
#define HAVE_TDM 1
#endif
#endif
#ifndef HAVE_TDM
#define HAVE_TDM 0
#endif

union FragU {
    unsigned u[8];
    uint4    q[2];
    v16bf    f;
};

__device__ inline v8f wmma_bf16(v16bf a, v16bf b, v8f c) {
    // D(16x16 f32) = A(16x32 bf16) * B(32x16 bf16) + C
    return __builtin_amdgcn_wmma_f32_16x16x32_bf16(
        /*neg_a=*/false, a, /*neg_b=*/false, b,
        /*c_mod=*/(short)0, c, /*reuse_a=*/false, /*reuse_b=*/false);
}

// A fragment (16x32 bf16) for this lane from a row-major LDS matrix.
// gfx1250 16-bit A layout: lanes 0-15 hold K={0..7,16..23} of row M=lane,
// lanes 16-31 hold K={8..15,24..31}. Pairs contiguous -> compiler emits 2x ds_load_b128.
__device__ inline v16bf frag_a_from_row(const __bf16* rowp, int kbase, int khalf, bool valid) {
    const unsigned* rp = (const unsigned*)rowp;   // 2 bf16 per dword
    FragU t;
#pragma unroll
    for (int v = 0; v < 8; ++v) {
        int k = kbase + ((v < 4) ? (khalf * 8 + 2 * v)
                                 : (16 + khalf * 8 + 2 * (v - 4)));
        t.u[v] = valid ? rp[k >> 1] : 0u;
    }
    return t.f;
}

// B fragments pre-packed in global memory as the per-lane register image:
// frag = 32 lanes x 8 dwords (32B/lane) -> two coalesced b128 loads.
__device__ inline v16bf frag_b_load(const unsigned* fragbase, int lane) {
    const uint4* q = (const uint4*)fragbase;
    FragU t;
    t.q[0] = q[lane * 2 + 0];
    t.q[1] = q[lane * 2 + 1];
    return t.f;
}

// ---------------------------------------------------------------------------
// Kernel 1: spectral norm (power iteration) + 10 Newton-Schulz iterations on
// the 64x64 basis. ~10 MFLOP total: one workgroup, VALU + LDS.
// ---------------------------------------------------------------------------
__global__ __launch_bounds__(256)
void ortho_kernel(const float* __restrict__ basis, float* __restrict__ basisO) {
    __shared__ float W[4096];
    __shared__ float T[4096];
    __shared__ float vvec[64], uvec[64];
    __shared__ float snorm;
    const int t = threadIdx.x;

    for (int i = t; i < 4096; i += 256) W[i] = basis[i];
    if (t < 64) vvec[t] = 1.0f;
    __syncthreads();

    for (int it = 0; it < 48; ++it) {
        if (t < 64) { float s = 0.f; for (int k = 0; k < 64; ++k) s += W[t*64+k]*vvec[k]; uvec[t] = s; }
        __syncthreads();
        if (t < 64) { float s = 0.f; for (int k = 0; k < 64; ++k) s += W[k*64+t]*uvec[k]; T[t] = s; }
        __syncthreads();
        if (t == 0) { float nn = 0.f; for (int k = 0; k < 64; ++k) nn += T[k]*T[k]; snorm = rsqrtf(nn + 1e-30f); }
        __syncthreads();
        if (t < 64) vvec[t] = T[t] * snorm;
        __syncthreads();
    }
    if (t < 64) { float s = 0.f; for (int k = 0; k < 64; ++k) s += W[t*64+k]*vvec[k]; uvec[t] = s; }
    __syncthreads();
    if (t == 0) { float nn = 0.f; for (int k = 0; k < 64; ++k) nn += uvec[k]*uvec[k]; snorm = 1.0f / sqrtf(nn + 1e-30f); }
    __syncthreads();
    for (int i = t; i < 4096; i += 256) W[i] *= snorm;   // W = M / sigma_max
    __syncthreads();

    for (int it = 0; it < 10; ++it) {                    // W <- 1.5W - 0.5 W (W^T W)
        for (int e = 0; e < 16; ++e) {
            int idx = t * 16 + e, i = idx >> 6, j = idx & 63;
            float s = 0.f;
            for (int k = 0; k < 64; ++k) s += W[k*64+i] * W[k*64+j];
            T[idx] = s;
        }
        __syncthreads();
        float wn[16];
        for (int e = 0; e < 16; ++e) {
            int idx = t * 16 + e, i = idx >> 6, j = idx & 63;
            float s = 0.f;
            for (int k = 0; k < 64; ++k) s += W[i*64+k] * T[k*64+j];
            wn[e] = 1.5f * W[idx] - 0.5f * s;
        }
        __syncthreads();
        for (int e = 0; e < 16; ++e) W[t*16+e] = wn[e];
        __syncthreads();
    }
    for (int i = t; i < 4096; i += 256) basisO[i] = W[i];
}

// ---------------------------------------------------------------------------
// Kernel 2: pack conv weights into WMMA B-fragment register images (bf16).
//   f in [0,8):    conv1  f = kc*4+nt               K = im2col(ic,ky,kx), 36->64 pad
//   f in [8,80):   conv2  f = 8  + (tap*2+kc)*4+nt  K = ic, tap = ky*3+kx
//   f in [80,208): conv3  f = 80 + (tap*2+kc)*4+nt  K = ic, tap = ky*4+kx
// B lane layout: n = lane&15, kh = lane>>4, vgpr v<4 holds K = 8kh+2v,+1; v>=4 holds K = 16+8kh+2(v-4),+1.
// ---------------------------------------------------------------------------
__global__ __launch_bounds__(256)
void prep_frags(const float* __restrict__ w1, const float* __restrict__ w2,
                const float* __restrict__ w3, unsigned* __restrict__ frags) {
    const int NF = 8 + 72 + 128;
    int gid = blockIdx.x * blockDim.x + threadIdx.x;
    if (gid >= NF * 32) return;
    int f = gid >> 5, lane = gid & 31;
    int n = lane & 15, kh = lane >> 4;

    unsigned o[8];
#pragma unroll
    for (int v = 0; v < 8; ++v) {
        unsigned pk = 0;
        for (int h = 0; h < 2; ++h) {
            int kl = ((v < 4) ? (kh * 8 + 2 * v) : (16 + kh * 8 + 2 * (v - 4))) + h;   // local K in [0,32)
            float val = 0.f;
            if (f < 8) {
                int kc = f >> 2, nt = f & 3;
                int k = kc * 32 + kl, oc = nt * 16 + n;
                if (k < 36) {
                    int ic = k / 9, r = (k % 9) / 3, c = k % 3;
                    val = w1[((oc * 4 + ic) * 3 + r) * 3 + c];
                }
            } else if (f < 80) {
                int g = f - 8;
                int tap = g >> 3, kc = (g >> 2) & 1, nt = g & 3;
                int ic = kc * 32 + kl, oc = nt * 16 + n;
                val = w2[((oc * 64 + ic) * 3 + tap / 3) * 3 + tap % 3];
            } else {
                int g = f - 80;
                int tap = g >> 3, kc = (g >> 2) & 1, nt = g & 3;
                int ic = kc * 32 + kl, oc = nt * 16 + n;
                val = w3[((oc * 64 + ic) * 4 + (tap >> 2)) * 4 + (tap & 3)];
            }
            __bf16 bv = (__bf16)val;
            unsigned short us = *(unsigned short*)&bv;
            pk |= ((unsigned)us) << (16 * h);
        }
        o[v] = pk;
    }
    unsigned* dst = frags + f * 256 + lane * 8;
#pragma unroll
    for (int v = 0; v < 8; ++v) dst[v] = o[v];
}

// ---------------------------------------------------------------------------
// Kernel 3: fused per-image pipeline, one workgroup (8 waves) per image.
//   TDM x-tile -> LDS, unshuffle+im2col -> conv1 WMMA -> conv2 WMMA (9 taps)
//   -> conv3 WMMA (16 taps) -> basis contraction. Activations live in LDS,
//   weights stream from L2-resident fragments.
// ---------------------------------------------------------------------------
__global__ __launch_bounds__(256)
void fused_main(const float* __restrict__ x,
                const float* __restrict__ b1, const float* __restrict__ b2,
                const float* __restrict__ b3, const float* __restrict__ basisO,
                const unsigned* __restrict__ w1f, const unsigned* __restrict__ w2f,
                const unsigned* __restrict__ w3f, float* __restrict__ out) {
    __shared__ float  xL[484];
    __shared__ __bf16 A1[128 * 64];   // conv1 im2col, K padded 36->64
    __shared__ __bf16 F1[128 * 64];   // conv1 output [pos][oc]
    __shared__ __bf16 F2[128 * 64];   // conv2 output [pos][oc]
    __shared__ float  CO[64 * 64];    // conv3 output [pos][oc] (f32)
    __shared__ float  RED[256];       // recon partial sums

    const int t = threadIdx.x;
    const int lane = t & 31;
    const int wave = t >> 5;
    const int img = blockIdx.x;

    __builtin_prefetch(w2f + wave * 2048, 0, 0);
    __builtin_prefetch(w3f + wave * 4096, 0, 0);

    const float* xin = x + img * 484;

#if HAVE_TDM
    // Tensor Data Mover: 1-D DMA of the 484-float tile into LDS (D# per ISA §8).
    if (wave == 0) {
        unsigned long long ga = (unsigned long long)(const void*)xin;
        v4u g0;
        g0[0] = 1u;                                      // count=1, user mode
        g0[1] = (unsigned)(size_t)&xL[0];                // lds_addr (flat low 32 = LDS offset)
        g0[2] = (unsigned)(ga & 0xFFFFFFFFu);            // global_addr[31:0]
        g0[3] = (unsigned)((ga >> 32) & 0x01FFFFFFu) | (2u << 30); // addr[56:32] | type=2
        v8i g1;
        g1[0] = (int)(2u << 16);                         // data_size = 4B
        g1[1] = (int)((484u & 0xFFFFu) << 16);           // tensor_dim0[15:0]
        g1[2] = (int)(1u << 16);                         // tensor_dim1 = 1
        g1[3] = (int)(484u << 16);                       // tile_dim0 = 484
        g1[4] = 1;                                       // tile_dim1 = 1
        g1[5] = 484;                                     // tensor_dim0_stride
        g1[6] = 0; g1[7] = 0;
        v4i gz = {0, 0, 0, 0};
#if __clang_major__ >= 23
        v8i gz8 = {0, 0, 0, 0, 0, 0, 0, 0};
        __builtin_amdgcn_tensor_load_to_lds(g0, g1, gz, gz, gz8, 0);
#else
        __builtin_amdgcn_tensor_load_to_lds(g0, g1, gz, gz, 0);
#endif
        __builtin_amdgcn_s_wait_tensorcnt(0);
    }
    __syncthreads();
#else
    for (int i = t; i < 484; i += 256) xL[i] = xin[i];
    __syncthreads();
#endif

    // pixel-unshuffle + im2col for conv1: A1[p][k], k = ic*9 + ky*3 + kx,
    // feat[ic][h][w] = x[2h+di][2w+dj], ic = di*2+dj (torch PixelUnshuffle order)
    for (int idx = t; idx < 128 * 64; idx += 256) {
        int p = idx >> 6, k = idx & 63;
        float val = 0.f;
        if (p < 121 && k < 36) {
            int y = p / 11, xx = p % 11;
            int ic = k / 9, r = (k % 9) / 3, c = k % 3;
            int h = y + r - 1, w = xx + c - 1;
            if ((unsigned)h < 11u && (unsigned)w < 11u)
                val = xL[(2 * h + (ic >> 1)) * 22 + (2 * w + (ic & 1))];
        }
        A1[idx] = (__bf16)val;
    }
    __syncthreads();

    // ---- conv1: M=128(121) x N=64 x K=64, wave = M-tile ----
    {
        const int mt = wave;
        const int row = mt * 16 + (lane & 15);
        const int kh = lane >> 4;
#pragma unroll
        for (int nt = 0; nt < 4; ++nt) {
            int col = nt * 16 + (lane & 15);
            float bv = b1[col];
            v8f acc = {bv, bv, bv, bv, bv, bv, bv, bv};  // bias folded into C
#pragma unroll
            for (int kc = 0; kc < 2; ++kc) {
                v16bf a = frag_a_from_row(&A1[row * 64], kc * 32, kh, true);
                v16bf b = frag_b_load(w1f + (kc * 4 + nt) * 256, lane);
                acc = wmma_bf16(a, b, acc);
            }
#pragma unroll
            for (int v = 0; v < 8; ++v) {
                int m = v + 8 * (lane >> 4);
                F1[(mt * 16 + m) * 64 + col] = (__bf16)fmaxf(acc[v], 0.f);
            }
        }
    }
    __syncthreads();

    // ---- conv2: 9 taps of M=128 x N=64 x K=64, wave = M-tile ----
    {
        const int mt = wave;
        const int p = mt * 16 + (lane & 15);
        const int kh = lane >> 4;
        const int py0 = p / 11, px0 = p % 11;
        v8f acc[4];
#pragma unroll
        for (int nt = 0; nt < 4; ++nt) {
            float bv = b2[nt * 16 + (lane & 15)];
            acc[nt] = (v8f){bv, bv, bv, bv, bv, bv, bv, bv};
        }
        for (int tap = 0; tap < 9; ++tap) {
            int yy = py0 + tap / 3 - 1, xx = px0 + tap % 3 - 1;
            bool valid = (p < 121) && ((unsigned)yy < 11u) && ((unsigned)xx < 11u);
            int srcp = valid ? (yy * 11 + xx) : 0;
#pragma unroll
            for (int kc = 0; kc < 2; ++kc) {
                v16bf a = frag_a_from_row(&F1[srcp * 64], kc * 32, kh, valid);
                const unsigned* wb = w2f + (tap * 2 + kc) * 4 * 256;
#pragma unroll
                for (int nt = 0; nt < 4; ++nt)
                    acc[nt] = wmma_bf16(a, frag_b_load(wb + nt * 256, lane), acc[nt]);
            }
        }
#pragma unroll
        for (int nt = 0; nt < 4; ++nt) {
            int col = nt * 16 + (lane & 15);
#pragma unroll
            for (int v = 0; v < 8; ++v) {
                int m = v + 8 * (lane >> 4);
                F2[(mt * 16 + m) * 64 + col] = (__bf16)fmaxf(acc[nt][v], 0.f);
            }
        }
    }
    __syncthreads();

    // ---- conv3 (4x4 valid, 11->8): 16 taps of M=64 x N=64 x K=64 ----
    // wave = (M-tile 0..3) x (N-half 0..1)
    {
        const int mt = wave & 3;
        const int nh = wave >> 2;
        const int q = mt * 16 + (lane & 15);
        const int kh = lane >> 4;
        const int qy = q >> 3, qx = q & 7;
        v8f acc[2];
#pragma unroll
        for (int j = 0; j < 2; ++j) {
            float bv = b3[(nh * 2 + j) * 16 + (lane & 15)];
            acc[j] = (v8f){bv, bv, bv, bv, bv, bv, bv, bv};
        }
        for (int tap = 0; tap < 16; ++tap) {
            int srcp = (qy + (tap >> 2)) * 11 + (qx + (tap & 3));
#pragma unroll
            for (int kc = 0; kc < 2; ++kc) {
                v16bf a = frag_a_from_row(&F2[srcp * 64], kc * 32, kh, true);
#pragma unroll
                for (int j = 0; j < 2; ++j) {
                    int nt = nh * 2 + j;
                    v16bf b = frag_b_load(w3f + ((tap * 2 + kc) * 4 + nt) * 256, lane);
                    acc[j] = wmma_bf16(a, b, acc[j]);
                }
            }
        }
#pragma unroll
        for (int j = 0; j < 2; ++j) {
            int col = (nh * 2 + j) * 16 + (lane & 15);
#pragma unroll
            for (int v = 0; v < 8; ++v) {
                int m = v + 8 * (lane >> 4);
                CO[(mt * 16 + m) * 64 + col] = acc[j][v];
            }
        }
    }
    __syncthreads();

    // ---- recon[q] = sum_oc basisO[oc][q] * coeff[q][oc], all 8 waves ----
    {
        int q = t & 63, part = t >> 6;        // 4 partials of 16 channels each
        float s = 0.f;
#pragma unroll
        for (int e = 0; e < 16; ++e) {
            int oc = part * 16 + e;
            s += basisO[oc * 64 + q] * CO[q * 64 + oc];
        }
        RED[t] = s;
    }
    __syncthreads();
    if (t < 64)
        out[img * 64 + t] = RED[t] + RED[t + 64] + RED[t + 128] + RED[t + 192];
}

// ---------------------------------------------------------------------------
extern "C" void kernel_launch(void* const* d_in, const int* in_sizes, int n_in,
                              void* d_out, int out_size, void* d_ws, size_t ws_size,
                              hipStream_t stream) {
    (void)in_sizes; (void)n_in; (void)out_size; (void)ws_size;
    const float* x     = (const float*)d_in[0];
    const float* w1    = (const float*)d_in[1];
    const float* b1    = (const float*)d_in[2];
    const float* w2    = (const float*)d_in[3];
    const float* b2    = (const float*)d_in[4];
    const float* w3    = (const float*)d_in[5];
    const float* b3    = (const float*)d_in[6];
    const float* basis = (const float*)d_in[7];
    float* out = (float*)d_out;

    // Workspace: [0,16K) basisO f32[4096]; [16K, 16K+208KB) 208 B-fragments.
    float*    basisO = (float*)d_ws;
    unsigned* frags  = (unsigned*)((char*)d_ws + 16384);

    ortho_kernel<<<1, 256, 0, stream>>>(basis, basisO);
    prep_frags<<<(208 * 32 + 255) / 256, 256, 0, stream>>>(w1, w2, w3, frags);
    fused_main<<<4096, 256, 0, stream>>>(x, b1, b2, b3, basisO,
                                         frags,             // conv1: frags 0..7
                                         frags + 8 * 256,   // conv2: frags 8..79
                                         frags + 80 * 256,  // conv3: frags 80..207
                                         out);
}
